// LinformerAttention_86002425135412
// MI455X (gfx1250) — compile-verified, matrix operand order checked by reading, not run on hardware
//
#include <hip/hip_runtime.h>

#define D_MODEL 1024
#define SEQ_LEN 4096
#define N_HEADS 16
#define HEAD_DIM 64
#define LKF 128
#define BATCH 4
#define M_TOTAL (BATCH * SEQ_LEN) /* 16384 */
#define SPLITL 8                  /* split-K chunks for kernel 2 */

typedef __attribute__((ext_vector_type(16))) _Float16 v16h;
typedef __attribute__((ext_vector_type(8)))  _Float16 v8h;
typedef __attribute__((ext_vector_type(8)))  float    v8f;

union V16H { v16h v; v8h h[2]; };

// Load one 16x32 f16 fragment for a lane.
// Per ISA 16-bit A/B layout: lane (0-15 -> m=lane, 16-31 -> m=lane-16) holds two
// contiguous 8-half runs: k in [kf, kf+8) and [kf+16, kf+24), kf = (lane>=16)?8:0.
__device__ __forceinline__ v16h frag_ld(const _Float16* base, int row, int stride, int koff) {
  V16H u;
  const _Float16* p = base + row * stride + koff;
  u.h[0] = *(const v8h*)(p);
  u.h[1] = *(const v8h*)(p + 16);
  return u.v;
}

__device__ __forceinline__ v8f wmma16(v16h a, v16h b, v8f c) {
  return __builtin_amdgcn_wmma_f32_16x16x32_f16(false, a, false, b, (short)0, c, false, false);
}

// ---------------------------------------------------------------------------
// Kernel 1: fused QKV projection.  C[m,n] = sum_k x[m,k] * W[n,k] + b[n]
// WG tile: 128(M) x 64(N) for each of Q,K,V, sharing the staged x tile.
// B-fragments are software-pipelined: frag t+1's ds_loads issue before WMMA t.
// ---------------------------------------------------------------------------
__launch_bounds__(256)
__global__ void qkv_proj_kernel(const float* __restrict__ x,
                                const float* __restrict__ Wq, const float* __restrict__ bq,
                                const float* __restrict__ Wk, const float* __restrict__ bk,
                                const float* __restrict__ Wv, const float* __restrict__ bv,
                                _Float16* __restrict__ Qh, _Float16* __restrict__ Kh,
                                _Float16* __restrict__ Vh)
{
  __shared__ _Float16 Al[128 * 40];      // x tile  [m][k], +8 pad
  __shared__ _Float16 Bl[3][64 * 40];    // W tiles [n][k]

  const int tid  = threadIdx.x;
  const int wave = tid >> 5;
  const int lane = tid & 31;
  const int mBase = blockIdx.x * 128;
  const int nBase = blockIdx.y * 64;

  const float* W[3]    = {Wq, Wk, Wv};
  const float* bias[3] = {bq, bk, bv};
  _Float16*    out[3]  = {Qh, Kh, Vh};

  v8f acc[3][4] = {};

  const int lr = tid >> 3;        // 0..31
  const int lc = (tid & 7) * 4;   // 0,4,...,28
  const int fr = lane & 15;
  const int kf = (lane >> 4) * 8;

  for (int k0 = 0; k0 < D_MODEL; k0 += 32) {
    __syncthreads();
    // stage x tile (128x32 fp32 -> f16)
#pragma unroll
    for (int rr = 0; rr < 4; ++rr) {
      const int row = lr + rr * 32;
      if (k0 + 32 < D_MODEL)
        __builtin_prefetch(x + (size_t)(mBase + row) * D_MODEL + k0 + 32 + lc, 0, 1);
      const float4 f = *(const float4*)(x + (size_t)(mBase + row) * D_MODEL + k0 + lc);
      _Float16* d = &Al[row * 40 + lc];
      d[0] = (_Float16)f.x; d[1] = (_Float16)f.y; d[2] = (_Float16)f.z; d[3] = (_Float16)f.w;
    }
    // stage Wq/Wk/Wv tiles (64x32 each, row-major along k already)
#pragma unroll
    for (int w = 0; w < 3; ++w) {
#pragma unroll
      for (int rr = 0; rr < 2; ++rr) {
        const int row = lr + rr * 32;
        const float4 f = *(const float4*)(W[w] + (size_t)(nBase + row) * D_MODEL + k0 + lc);
        _Float16* d = &Bl[w][row * 40 + lc];
        d[0] = (_Float16)f.x; d[1] = (_Float16)f.y; d[2] = (_Float16)f.z; d[3] = (_Float16)f.w;
      }
    }
    __syncthreads();

    const v16h a = frag_ld(&Al[(wave * 16) * 40], fr, 40, kf);
    // 12-deep pipelined WMMA chain over {w, t}
    v16h bc = frag_ld(&Bl[0][0], fr, 40, kf);
#pragma unroll
    for (int i = 0; i < 11; ++i) {
      const int j = i + 1;
      const v16h bn = frag_ld(&Bl[j >> 2][((j & 3) * 16) * 40], fr, 40, kf);
      acc[i >> 2][i & 3] = wmma16(a, bc, acc[i >> 2][i & 3]);
      bc = bn;
    }
    acc[2][3] = wmma16(a, bc, acc[2][3]);
  }

  // epilogue: bias add, f32 -> f16 store.  C layout: vgpr r -> m = r (lanes 0-15)
  // or r+8 (lanes 16-31); n = lane&15 within the 16-col tile.
  const int col  = lane & 15;
  const int rofs = (lane >> 4) * 8;
#pragma unroll
  for (int w = 0; w < 3; ++w) {
#pragma unroll
    for (int t = 0; t < 4; ++t) {
      const int n = nBase + t * 16 + col;
      const float bb = bias[w][n];
#pragma unroll
      for (int r = 0; r < 8; ++r) {
        const int m = mBase + wave * 16 + rofs + r;
        out[w][(size_t)m * D_MODEL + n] = (_Float16)(acc[w][t][r] + bb);
      }
    }
  }
}

// ---------------------------------------------------------------------------
// Kernel 2: Kp/Vp partials.  Kp[k,d] = sum_l E[h,k,l] * K[b,l,h*64+d]
// grid (64 bh, SPLITL);  each WG reduces a 512-wide l-chunk -> partial buffer.
// ---------------------------------------------------------------------------
__launch_bounds__(256)
__global__ void kv_lowrank_kernel(const float* __restrict__ E,
                                  const _Float16* __restrict__ Kh,
                                  const _Float16* __restrict__ Vh,
                                  float* __restrict__ KpPart, float* __restrict__ VpPart)
{
  __shared__ _Float16 Al[128 * 40];   // E tile  [krow][l]
  __shared__ _Float16 BK[64 * 40];    // K^T tile [d][l]
  __shared__ _Float16 BV[64 * 40];

  const int tid  = threadIdx.x;
  const int wave = tid >> 5;
  const int lane = tid & 31;
  const int bh = blockIdx.x;          // 0..63
  const int b  = bh >> 4, h = bh & 15;
  const int ls = blockIdx.y;          // split index

  v8f accK[4] = {}, accV[4] = {};

  const int lr = tid >> 3, lc = (tid & 7) * 4;
  const int tl = tid >> 3;            // l within tile for B stage
  const int td = (tid & 7) * 8;       // d group
  const int fr = lane & 15;
  const int kf = (lane >> 4) * 8;

  const size_t eBase  = (size_t)h * LKF * SEQ_LEN;
  const size_t kvBase = (size_t)b * SEQ_LEN * D_MODEL + (size_t)h * HEAD_DIM;
  const int lBeg = ls * (SEQ_LEN / SPLITL);
  const int lEnd = lBeg + (SEQ_LEN / SPLITL);

  for (int l0 = lBeg; l0 < lEnd; l0 += 32) {
    __syncthreads();
#pragma unroll
    for (int rr = 0; rr < 4; ++rr) {
      const int row = lr + rr * 32;
      const float4 f = *(const float4*)(E + eBase + (size_t)row * SEQ_LEN + l0 + lc);
      _Float16* d = &Al[row * 40 + lc];
      d[0] = (_Float16)f.x; d[1] = (_Float16)f.y; d[2] = (_Float16)f.z; d[3] = (_Float16)f.w;
    }
    {
      const size_t src = kvBase + (size_t)(l0 + tl) * D_MODEL + td;
      const v8h kk = *(const v8h*)(Kh + src);
      const v8h vv = *(const v8h*)(Vh + src);
#pragma unroll
      for (int i = 0; i < 8; ++i) { BK[(td + i) * 40 + tl] = kk[i]; BV[(td + i) * 40 + tl] = vv[i]; }
    }
    __syncthreads();

    const v16h a = frag_ld(&Al[(wave * 16) * 40], fr, 40, kf);
    // pipelined over t: next K/V fragments in flight during current WMMAs
    v16h bkc = frag_ld(&BK[0], fr, 40, kf);
    v16h bvc = frag_ld(&BV[0], fr, 40, kf);
#pragma unroll
    for (int t = 0; t < 3; ++t) {
      const v16h bkn = frag_ld(&BK[((t + 1) * 16) * 40], fr, 40, kf);
      const v16h bvn = frag_ld(&BV[((t + 1) * 16) * 40], fr, 40, kf);
      accK[t] = wmma16(a, bkc, accK[t]);
      accV[t] = wmma16(a, bvc, accV[t]);
      bkc = bkn; bvc = bvn;
    }
    accK[3] = wmma16(a, bkc, accK[3]);
    accV[3] = wmma16(a, bvc, accV[3]);
  }

  const int col = lane & 15, rofs = (lane >> 4) * 8;
  const size_t pb = ((size_t)ls * (BATCH * N_HEADS) + bh) * LKF * HEAD_DIM;
#pragma unroll
  for (int t = 0; t < 4; ++t) {
#pragma unroll
    for (int r = 0; r < 8; ++r) {
      const int krow = wave * 16 + rofs + r;
      const int n = t * 16 + col;
      KpPart[pb + (size_t)krow * HEAD_DIM + n] = accK[t][r];
      VpPart[pb + (size_t)krow * HEAD_DIM + n] = accV[t][r];
    }
  }
}

// Reduce SPLITL partials -> f16 Kp/Vp (deterministic, no atomics).
__launch_bounds__(256)
__global__ void kv_reduce_kernel(const float* __restrict__ KpPart,
                                 const float* __restrict__ VpPart,
                                 _Float16* __restrict__ Kp, _Float16* __restrict__ Vp)
{
  const size_t idx = (size_t)blockIdx.x * 256 + threadIdx.x;  // 0..524287
  const size_t stride = (size_t)BATCH * N_HEADS * LKF * HEAD_DIM;
  float sk = 0.f, sv = 0.f;
#pragma unroll
  for (int s = 0; s < SPLITL; ++s) { sk += KpPart[s * stride + idx]; sv += VpPart[s * stride + idx]; }
  Kp[idx] = (_Float16)sk;
  Vp[idx] = (_Float16)sv;
}

// ---------------------------------------------------------------------------
// Kernel 3: attention.  dot = Q @ Kp^T / 8 -> softmax(k=128) -> @ Vp -> out.
// grid (32 l-blocks, 64 bh); WG covers 128 rows of l (16 per wave).
// LDS: [VpL 64x136][ KpL 128x72  aliased with  AtL 8x16x136 ]  (52 KB)
// ---------------------------------------------------------------------------
__launch_bounds__(256)
__global__ void attn_kernel(const _Float16* __restrict__ Qh,
                            const _Float16* __restrict__ Kp, const _Float16* __restrict__ Vp,
                            float* __restrict__ out)
{
  __shared__ _Float16 smem[64 * 136 + 8 * 16 * 136];
  _Float16* VpL = smem;                 // [d][kidx]  (transposed Vp)
  _Float16* KpL = smem + 64 * 136;      // [kidx][d]  (row-major Kp) -- dead after GEMM1
  _Float16* AtL = smem + 64 * 136;      // [wave][m][kidx] attn rows -- reuses KpL space

  const int tid  = threadIdx.x;
  const int wave = tid >> 5;
  const int lane = tid & 31;
  const int lBase = blockIdx.x * 128;
  const int bh = blockIdx.y;
  const int b  = bh >> 4, h = bh & 15;

  // stage Kp (row-major) and Vp (transposed)
  {
    const int kr = tid >> 1;
    const int c0 = (tid & 1) * 32;
    const size_t src = ((size_t)bh * LKF + kr) * HEAD_DIM + c0;
#pragma unroll
    for (int j = 0; j < 4; ++j) {
      const v8h kk = *(const v8h*)(Kp + src + 8 * j);
      *(v8h*)(&KpL[kr * 72 + c0 + 8 * j]) = kk;
      const v8h vv = *(const v8h*)(Vp + src + 8 * j);
#pragma unroll
      for (int i = 0; i < 8; ++i) VpL[(c0 + 8 * j + i) * 136 + kr] = vv[i];
    }
  }

  const int fr = lane & 15;
  const int kf = (lane >> 4) * 8;
  const int rofs = (lane >> 4) * 8, col = lane & 15;

  // hoist both global Q fragments (k-halves 0..31, 32..63) before the barrier
  const _Float16* qBase = Qh + ((size_t)b * SEQ_LEN + lBase + wave * 16) * D_MODEL
                             + (size_t)h * HEAD_DIM;
  v16h afr[2];
#pragma unroll
  for (int ks = 0; ks < 2; ++ks) afr[ks] = frag_ld(qBase, fr, D_MODEL, ks * 32 + kf);

  __syncthreads();

  // GEMM1: dot[l, kidx] = Q[l,:64] . Kp[kidx,:64], pipelined over the 8 n-tiles
  v8f acc[8] = {};
#pragma unroll
  for (int ks = 0; ks < 2; ++ks) {
    v16h bc = frag_ld(&KpL[0], fr, 72, ks * 32 + kf);
#pragma unroll
    for (int t = 0; t < 7; ++t) {
      const v16h bn = frag_ld(&KpL[((t + 1) * 16) * 72], fr, 72, ks * 32 + kf);
      acc[t] = wmma16(afr[ks], bc, acc[t]);
      bc = bn;
    }
    acc[7] = wmma16(afr[ks], bc, acc[7]);
  }
  __syncthreads();   // KpL dead everywhere; AtL may now reuse its space

  // softmax over 128 per row; rows split across half-waves, reduce over 16 lanes
  const float scale = 0.125f;  // 1/sqrt(64)
  _Float16* at = &AtL[(wave * 16) * 136];
#pragma unroll
  for (int r = 0; r < 8; ++r) {
    float mx = -3.0e38f;
#pragma unroll
    for (int t = 0; t < 8; ++t) { acc[t][r] *= scale; mx = fmaxf(mx, acc[t][r]); }
#pragma unroll
    for (int s = 1; s < 16; s <<= 1) mx = fmaxf(mx, __shfl_xor(mx, s, 32));
    float sum = 0.f;
#pragma unroll
    for (int t = 0; t < 8; ++t) { const float e = __expf(acc[t][r] - mx); acc[t][r] = e; sum += e; }
#pragma unroll
    for (int s = 1; s < 16; s <<= 1) sum += __shfl_xor(sum, s, 32);
    const float inv = 1.0f / sum;
    const int m = rofs + r;
#pragma unroll
    for (int t = 0; t < 8; ++t) at[m * 136 + t * 16 + col] = (_Float16)(acc[t][r] * inv);
  }
  __syncthreads();

  // GEMM2: out[l, d] = attn[l,:128] . Vp[:,d], pipelined over the 4 n-tiles
  v8f o[4] = {};
#pragma unroll
  for (int ks = 0; ks < 4; ++ks) {
    const v16h a = frag_ld(at, fr, 136, ks * 32 + kf);
    v16h bc = frag_ld(&VpL[0], fr, 136, ks * 32 + kf);
#pragma unroll
    for (int t = 0; t < 3; ++t) {
      const v16h bn = frag_ld(&VpL[((t + 1) * 16) * 136], fr, 136, ks * 32 + kf);
      o[t] = wmma16(a, bc, o[t]);
      bc = bn;
    }
    o[3] = wmma16(a, bc, o[3]);
  }
  // epilogue: head-combined fp32 output [b, l, h*64 + d]
#pragma unroll
  for (int t = 0; t < 4; ++t) {
#pragma unroll
    for (int r = 0; r < 8; ++r) {
      const int l = lBase + wave * 16 + rofs + r;
      out[((size_t)b * SEQ_LEN + l) * D_MODEL + (size_t)h * HEAD_DIM + t * 16 + col] = o[t][r];
    }
  }
}

// ---------------------------------------------------------------------------
extern "C" void kernel_launch(void* const* d_in, const int* in_sizes, int n_in,
                              void* d_out, int out_size, void* d_ws, size_t ws_size,
                              hipStream_t stream) {
  const float* x  = (const float*)d_in[0];
  const float* Wq = (const float*)d_in[1];
  const float* bq = (const float*)d_in[2];
  const float* Wk = (const float*)d_in[3];
  const float* bk = (const float*)d_in[4];
  const float* Wv = (const float*)d_in[5];
  const float* bv = (const float*)d_in[6];
  const float* E  = (const float*)d_in[7];
  float* outp = (float*)d_out;

  char* ws = (char*)d_ws;
  const size_t szQKV  = (size_t)M_TOTAL * D_MODEL * sizeof(_Float16);          // 32 MB each
  const size_t szProj = (size_t)BATCH * N_HEADS * LKF * HEAD_DIM;              // 524288 elems
  _Float16* Qh = (_Float16*)(ws);
  _Float16* Kh = (_Float16*)(ws + szQKV);
  _Float16* Vh = (_Float16*)(ws + 2 * szQKV);
  float* KpPart = (float*)(ws + 3 * szQKV);                                    // 16 MB
  float* VpPart = (float*)(ws + 3 * szQKV + SPLITL * szProj * sizeof(float));  // 16 MB
  _Float16* Kp = (_Float16*)(ws + 3 * szQKV + 2 * SPLITL * szProj * sizeof(float));
  _Float16* Vp = (_Float16*)((char*)Kp + szProj * sizeof(_Float16));

  qkv_proj_kernel<<<dim3(M_TOTAL / 128, D_MODEL / 64), 256, 0, stream>>>(
      x, Wq, bq, Wk, bk, Wv, bv, Qh, Kh, Vh);

  kv_lowrank_kernel<<<dim3(BATCH * N_HEADS, SPLITL), 256, 0, stream>>>(
      E, Kh, Vh, KpPart, VpPart);

  kv_reduce_kernel<<<dim3((unsigned)(szProj / 256)), 256, 0, stream>>>(
      KpPart, VpPart, Kp, Vp);

  attn_kernel<<<dim3(SEQ_LEN / 128, BATCH * N_HEADS), 256, 0, stream>>>(
      Qh, Kp, Vp, outp);
}
